// QuantiZ_19035295056273
// MI455X (gfx1250) — matrix operand order repaired and checked by reading, not compiled, not run on hardware
//
#include <hip/hip_runtime.h>
#include <hip/hip_bf16.h>

// ---------------------------------------------------------------------------
// QuantiZ: zidx = argmin_k ||z - e_k||, quant = e[zidx]
//   e = codebook_w @ proj_w^T + proj_b        [4096, 256]
//   z = encode.reshape(-1, 256)               [32768, 256]
// argmax(softmax(-(cdist - MU)/SIGMA)) == argmax_k (z.e_k - 0.5*|e_k|^2)
// Heavy math in bf16 WMMA (v_wmma_f32_16x16x32_bf16), f32 accumulate.
// ---------------------------------------------------------------------------

#define NUM_CODE 4096
#define CODE_DIM 256
#define IN_DIM   1024
#define TOKENS   32768   // 8*64*64

typedef __attribute__((ext_vector_type(16))) __bf16 v16bf;
typedef __attribute__((ext_vector_type(8)))  __bf16 bf16x8;
typedef __attribute__((ext_vector_type(2)))  __bf16 bf16x2;
typedef __attribute__((ext_vector_type(8)))  float  v8f;

// pack two f32 -> two bf16 (RNE) in one dword, using HW convert when present
__device__ __forceinline__ unsigned pk2bf(float a, float b) {
  union { bf16x2 v; unsigned u; } c;
#if __has_builtin(__builtin_amdgcn_cvt_pk_bf16_f32)
  c.v = __builtin_amdgcn_cvt_pk_bf16_f32(a, b);
#else
  c.v.x = (__bf16)a; c.v.y = (__bf16)b;
#endif
  return c.u;
}

__device__ __forceinline__ unsigned short f2bf(float f) {
  unsigned u = __float_as_uint(f);
  u += 0x7FFFu + ((u >> 16) & 1u);
  return (unsigned short)(u >> 16);
}

__device__ __forceinline__ v8f wmma_bf16(v16bf a, v16bf b, v8f c) {
  return __builtin_amdgcn_wmma_f32_16x16x32_bf16(false, a, false, b,
                                                 (short)0, c, false, false);
}

// A-matrix 16x32 bf16 fragment from row-major LDS tile [rows][stride]
__device__ __forceinline__ v16bf load_frag_a(const unsigned short* lds,
                                             int rbase, int stride, int ko) {
  const int l  = threadIdx.x & 31;
  const int hf = l >> 4;
  const unsigned short* row = lds + (rbase + (l & 15)) * stride;
  union { v16bf v; bf16x8 h[2]; } u;
  u.h[0] = *(const bf16x8*)(row + ko + 8 * hf);
  u.h[1] = *(const bf16x8*)(row + ko + 16 + 8 * hf);
  return u.v;
}

// B-matrix 32x16 bf16 fragment, B[k][n] = tile[n][k] (row-major [n][stride])
__device__ __forceinline__ v16bf load_frag_b(const unsigned short* lds,
                                             int rbase, int stride, int ko) {
  const int l  = threadIdx.x & 31;
  const int hf = l >> 4;
  const unsigned short* row = lds + (rbase + (l & 15)) * stride;
  union { v16bf v; bf16x8 h[2]; } u;
  u.h[0] = *(const bf16x8*)(row + ko + 8 * hf);
  u.h[1] = *(const bf16x8*)(row + ko + 16 + 8 * hf);
  return u.v;
}

// ---------------------------------------------------------------------------
// K1: e = codebook @ proj^T + b   (M=4096, N=256, K=1024) via bf16 WMMA
// ---------------------------------------------------------------------------
__global__ __launch_bounds__(256) void k1_codebook_proj(
    const float* __restrict__ codebook, const float* __restrict__ projw,
    const float* __restrict__ projb, float* __restrict__ e_f32,
    unsigned short* __restrict__ e_bf16) {
  __shared__ alignas(16) unsigned short lA[128 * 64];
  __shared__ alignas(16) unsigned short lB[64 * 64];
  const int m0 = blockIdx.x * 128;
  const int n0 = blockIdx.y * 64;
  const int tid = threadIdx.x, wave = tid >> 5, lane = tid & 31;
  const int r16 = tid >> 4, c4 = tid & 15;   // 16 float4 per 64-wide row

  v8f acc[4];
  #pragma unroll
  for (int i = 0; i < 4; ++i) acc[i] = (v8f){0.f,0.f,0.f,0.f,0.f,0.f,0.f,0.f};

  for (int kc = 0; kc < IN_DIM; kc += 64) {
    #pragma unroll
    for (int i = 0; i < 8; ++i) {               // A: 128x64 f32 -> bf16
      int row = r16 + i * 16;
      float4 f = ((const float4*)(codebook + (size_t)(m0 + row) * IN_DIM + kc))[c4];
      unsigned* d = (unsigned*)(lA + row * 64 + c4 * 4);
      d[0] = pk2bf(f.x, f.y); d[1] = pk2bf(f.z, f.w);
    }
    #pragma unroll
    for (int i = 0; i < 4; ++i) {               // B: 64x64 f32 -> bf16
      int row = r16 + i * 16;
      float4 f = ((const float4*)(projw + (size_t)(n0 + row) * IN_DIM + kc))[c4];
      unsigned* d = (unsigned*)(lB + row * 64 + c4 * 4);
      d[0] = pk2bf(f.x, f.y); d[1] = pk2bf(f.z, f.w);
    }
    __syncthreads();
    #pragma unroll
    for (int ko = 0; ko < 64; ko += 32) {
      v16bf a  = load_frag_a(lA, wave * 16, 64, ko);
      v16bf b0 = load_frag_b(lB,  0, 64, ko);
      v16bf b1 = load_frag_b(lB, 16, 64, ko);
      v16bf b2 = load_frag_b(lB, 32, 64, ko);
      v16bf b3 = load_frag_b(lB, 48, 64, ko);
      acc[0] = wmma_bf16(a, b0, acc[0]);
      acc[1] = wmma_bf16(a, b1, acc[1]);
      acc[2] = wmma_bf16(a, b2, acc[2]);
      acc[3] = wmma_bf16(a, b3, acc[3]);
    }
    __syncthreads();
  }
  const int nl = lane & 15, hf = lane >> 4;
  #pragma unroll
  for (int ns = 0; ns < 4; ++ns) {
    int n = n0 + ns * 16 + nl;
    float bias = projb[n];
    #pragma unroll
    for (int r = 0; r < 8; ++r) {
      int m = m0 + wave * 16 + r + 8 * hf;      // C layout: VGPR r -> M=r(+8)
      float v = acc[ns][r] + bias;
      e_f32 [(size_t)m * CODE_DIM + n] = v;
      e_bf16[(size_t)m * CODE_DIM + n] = f2bf(v);
    }
  }
}

// ---------------------------------------------------------------------------
// K2: half_e2[k] = 0.5 * sum(e[k]^2)   one wave per row
// ---------------------------------------------------------------------------
__global__ __launch_bounds__(256) void k2_half_e2(
    const float* __restrict__ e_f32, float* __restrict__ half_e2) {
  const int wave = threadIdx.x >> 5, lane = threadIdx.x & 31;
  const int row = blockIdx.x * 8 + wave;
  const float* p = e_f32 + (size_t)row * CODE_DIM;
  float s = 0.f;
  #pragma unroll
  for (int i = 0; i < CODE_DIM / 32; ++i) { float v = p[lane + i * 32]; s += v * v; }
  #pragma unroll
  for (int off = 16; off; off >>= 1) s += __shfl_xor(s, off, 32);
  if (lane == 0) half_e2[row] = 0.5f * s;
}

// ---------------------------------------------------------------------------
// K3: scores + argmax + gather. 128 z-rows per block, 8 waves each owning a
// 16-row M-subtile. e-blocks (64 codes) double-buffered through registers.
// ---------------------------------------------------------------------------
__global__ __launch_bounds__(256) void k3_argmax_gather(
    const float* __restrict__ z, const unsigned short* __restrict__ e_bf16,
    const float* __restrict__ e_f32, const float* __restrict__ half_e2,
    int* __restrict__ zidx, float* __restrict__ quant) {
  __shared__ alignas(16) unsigned short zt[128 * 256];   // 64 KB
  __shared__ alignas(16) unsigned short et[64 * 256];    // 32 KB
  __shared__ int bestIdxSh[128];

  const int tid = threadIdx.x, wave = tid >> 5, lane = tid & 31;
  const int nl = lane & 15, hf = lane >> 4;
  const int t0 = blockIdx.x * 128;

  // --- stage z tile: 128x256 f32 -> bf16 (64 float4 per row) ---
  const int zr0 = tid >> 6, zc4 = tid & 63;
  #pragma unroll
  for (int i = 0; i < 32; ++i) {
    int row = zr0 + i * 4;
    float4 f = ((const float4*)(z + (size_t)(t0 + row) * CODE_DIM))[zc4];
    unsigned* d = (unsigned*)(zt + row * 256 + zc4 * 4);
    d[0] = pk2bf(f.x, f.y); d[1] = pk2bf(f.z, f.w);
  }

  // --- prime e-block register stage (block 0): 64x256 bf16 = 2048 uint4 ---
  const int er0 = tid >> 5, ec8 = tid & 31;   // 32 uint4 per 256-wide row
  uint4 stg[8];
  #pragma unroll
  for (int i = 0; i < 8; ++i)
    stg[i] = ((const uint4*)(e_bf16 + (size_t)(er0 + i * 8) * CODE_DIM))[ec8];

  float best[8]; int bidx[8];
  #pragma unroll
  for (int r = 0; r < 8; ++r) { best[r] = -3.4e38f; bidx[r] = 0; }

  for (int blk = 0; blk < NUM_CODE; blk += 64) {
    __syncthreads();                            // prior reads of et done
    #pragma unroll
    for (int i = 0; i < 8; ++i)                 // regs -> LDS
      ((uint4*)(et + (er0 + i * 8) * 256))[ec8] = stg[i];
    if (blk + 64 < NUM_CODE) {                  // fetch next block during compute
      #pragma unroll
      for (int i = 0; i < 8; ++i)
        stg[i] = ((const uint4*)(e_bf16 +
                  (size_t)(blk + 64 + er0 + i * 8) * CODE_DIM))[ec8];
    }
    __syncthreads();

    v8f acc[4];
    #pragma unroll
    for (int i = 0; i < 4; ++i) acc[i] = (v8f){0.f,0.f,0.f,0.f,0.f,0.f,0.f,0.f};
    #pragma unroll
    for (int ko = 0; ko < 256; ko += 32) {      // all frags loaded before WMMAs
      v16bf a  = load_frag_a(zt, wave * 16, 256, ko);
      v16bf b0 = load_frag_b(et,  0, 256, ko);
      v16bf b1 = load_frag_b(et, 16, 256, ko);
      v16bf b2 = load_frag_b(et, 32, 256, ko);
      v16bf b3 = load_frag_b(et, 48, 256, ko);
      acc[0] = wmma_bf16(a, b0, acc[0]);
      acc[1] = wmma_bf16(a, b1, acc[1]);
      acc[2] = wmma_bf16(a, b2, acc[2]);
      acc[3] = wmma_bf16(a, b3, acc[3]);
    }
    #pragma unroll
    for (int ns = 0; ns < 4; ++ns) {
      int n = blk + ns * 16 + nl;
      float he = half_e2[n];
      #pragma unroll
      for (int r = 0; r < 8; ++r) {
        float s = acc[ns][r] - he;              // z.e_n - 0.5|e_n|^2
        if (s > best[r]) { best[r] = s; bidx[r] = n; }  // first-max kept
      }
    }
  }

  #pragma unroll
  for (int r = 0; r < 8; ++r) {                 // reduce over the 16 N-lanes
    float b = best[r]; int i = bidx[r];
    #pragma unroll
    for (int off = 1; off < 16; off <<= 1) {
      float ob = __shfl_xor(b, off, 32);
      int   oi = __shfl_xor(i, off, 32);
      if (ob > b || (ob == b && oi < i)) { b = ob; i = oi; }
    }
    if (nl == 0) {
      int mrow = wave * 16 + r + 8 * hf;
      bestIdxSh[mrow] = i;
      zidx[t0 + mrow]  = i;
    }
  }
  __syncthreads();
  #pragma unroll
  for (int i = 0; i < 32; ++i) {                // quant = e_f32[idx] gather
    int row = zr0 + i * 4;
    int ci = bestIdxSh[row];
    ((float4*)(quant + (size_t)(t0 + row) * CODE_DIM))[zc4] =
        ((const float4*)(e_f32 + (size_t)ci * CODE_DIM))[zc4];
  }
}

// ---------------------------------------------------------------------------
extern "C" void kernel_launch(void* const* d_in, const int* in_sizes, int n_in,
                              void* d_out, int out_size, void* d_ws, size_t ws_size,
                              hipStream_t stream) {
  const float* encode   = (const float*)d_in[0];   // [8,64,64,256]
  const float* codebook = (const float*)d_in[1];   // [4096,1024]
  const float* projw    = (const float*)d_in[2];   // [256,1024]
  const float* projb    = (const float*)d_in[3];   // [256]

  char* ws = (char*)d_ws;
  float*          e_f32   = (float*)ws;                                      // 4 MB
  unsigned short* e_bf16  = (unsigned short*)(ws + (size_t)NUM_CODE * CODE_DIM * 4);
  float*          half_e2 = (float*)(ws + (size_t)NUM_CODE * CODE_DIM * 6);  // +16KB

  int*   zidx  = (int*)d_out;                      // 32768 int32
  float* quant = (float*)d_out + TOKENS;           // 32768*256 f32

  k1_codebook_proj<<<dim3(32, 4), 256, 0, stream>>>(codebook, projw, projb,
                                                    e_f32, e_bf16);
  k2_half_e2<<<512, 256, 0, stream>>>(e_f32, half_e2);
  k3_argmax_gather<<<TOKENS / 128, 256, 0, stream>>>(encode, e_bf16, e_f32,
                                                     half_e2, zidx, quant);
}
